// AFlowNet_1941325218208
// MI455X (gfx1250) — matrix operand order repaired and checked by reading, not compiled, not run on hardware
//
#include <hip/hip_runtime.h>
#include <math.h>

typedef __attribute__((ext_vector_type(16))) _Float16 v16h;
typedef __attribute__((ext_vector_type(8)))  float    v8f;

#define BATCH 2
#define FPN   256
#define IMG   256

// ---------------------------------------------------------------------------
// Bilinear grid sample. img (B,C,H,W), flow holds normalized grid as (B,2,H,W)
// (channel 0 = gx, channel 1 = gy). out (B,C,H,W).
// ---------------------------------------------------------------------------
__global__ void k_grid_sample(float* __restrict__ out, const float* __restrict__ img,
                              const float* __restrict__ flow, int C, int H, int W) {
  int idx = blockIdx.x * blockDim.x + threadIdx.x;
  int HW = H * W;
  int total = BATCH * C * HW;
  if (idx >= total) return;
  int x = idx % W;
  int y = (idx / W) % H;
  int c = (idx / HW) % C;
  int b = idx / (HW * C);
  float gxn = flow[(b * 2 + 0) * HW + y * W + x];
  float gyn = flow[(b * 2 + 1) * HW + y * W + x];
  float gx = (gxn + 1.0f) * 0.5f * (float)(W - 1);
  float gy = (gyn + 1.0f) * 0.5f * (float)(H - 1);
  gx = fminf(fmaxf(gx, 0.0f), (float)(W - 1));
  gy = fminf(fmaxf(gy, 0.0f), (float)(H - 1));
  float x0f = floorf(gx), y0f = floorf(gy);
  float wx = gx - x0f, wy = gy - y0f;
  int x0 = (int)x0f, y0 = (int)y0f;
  int x1 = x0 + 1 < W - 1 ? x0 + 1 : W - 1;
  int y1 = y0 + 1 < H - 1 ? y0 + 1 : H - 1;
  const float* ip = img + (size_t)(b * C + c) * HW;
  float v00 = ip[y0 * W + x0];
  float v01 = ip[y0 * W + x1];
  float v10 = ip[y1 * W + x0];
  float v11 = ip[y1 * W + x1];
  float top = v00 * (1.0f - wx) + v01 * wx;
  float bot = v10 * (1.0f - wx) + v11 * wx;
  out[idx] = top * (1.0f - wy) + bot * wy;
}

// ---------------------------------------------------------------------------
// 7x7 correlation (mean over 256 channels) + leaky_relu(0.1).
// ---------------------------------------------------------------------------
__global__ void k_corr_leaky(float* __restrict__ out, const float* __restrict__ first,
                             const float* __restrict__ second, int H, int W) {
  int idx = blockIdx.x * blockDim.x + threadIdx.x;
  int HW = H * W;
  if (idx >= BATCH * HW) return;
  int x = idx % W;
  int y = (idx / W) % H;
  int b = idx / HW;
  float acc[49];
#pragma unroll
  for (int k = 0; k < 49; ++k) acc[k] = 0.0f;
  const float* fp = first + (size_t)b * FPN * HW;
  const float* sp = second + (size_t)b * FPN * HW;
  for (int c = 0; c < FPN; ++c) {
    float f = fp[c * HW + y * W + x];
#pragma unroll
    for (int dy = 0; dy < 7; ++dy) {
      int yy = y + dy - 3;
#pragma unroll
      for (int dx = 0; dx < 7; ++dx) {
        int xx = x + dx - 3;
        float s = (yy >= 0 && yy < H && xx >= 0 && xx < W) ? sp[c * HW + yy * W + xx] : 0.0f;
        acc[dy * 7 + dx] += f * s;
      }
    }
  }
  const float inv = 1.0f / (float)FPN;
#pragma unroll
  for (int k = 0; k < 49; ++k) {
    float v = acc[k] * inv;
    out[(size_t)(b * 49 + k) * HW + y * W + x] = v > 0.0f ? v : 0.1f * v;
  }
}

// ---------------------------------------------------------------------------
// Depthwise 3x3 conv (SAME) + bias, with output written as WMMA-B-fragment
// packed f16, zero-padded in K to Cpad (multiple of 32).
// Packed layout (halves): [b][p_tile][k_chunk][lane 0..31][j 0..15]
//   lane = (K_in_chunk>=16)*16 + (pixel & 15);  j = K_in_chunk & 15
// Input may be a virtual concat of src0 (csplit ch) and src1 (C-csplit ch).
// ---------------------------------------------------------------------------
__global__ void k_dwconv_pack(_Float16* __restrict__ out, const float* __restrict__ src0,
                              const float* __restrict__ src1, int csplit, int C, int Cpad,
                              const float* __restrict__ w, const float* __restrict__ bias,
                              int H, int W) {
  int idx = blockIdx.x * blockDim.x + threadIdx.x;
  int HW = H * W;
  int total = BATCH * Cpad * HW;
  if (idx >= total) return;
  int x = idx % W;
  int y = (idx / W) % H;
  int c = (idx / HW) % Cpad;
  int b = idx / (HW * Cpad);
  float acc = 0.0f;
  if (c < C) {
    const float* ip;
    if (c < csplit) ip = src0 + (size_t)(b * csplit + c) * HW;
    else            ip = src1 + (size_t)(b * (C - csplit) + (c - csplit)) * HW;
    const float* wp = w + c * 9;
    acc = bias[c];
#pragma unroll
    for (int ky = 0; ky < 3; ++ky) {
      int yy = y + ky - 1;
#pragma unroll
      for (int kx = 0; kx < 3; ++kx) {
        int xx = x + kx - 1;
        float v = (yy >= 0 && yy < H && xx >= 0 && xx < W) ? ip[yy * W + xx] : 0.0f;
        acc += v * wp[ky * 3 + kx];
      }
    }
  }
  // packed write
  int p = y * W + x;
  int pt = p >> 4, lp = p & 15;
  int kc = c >> 5, rem = c & 31;
  int half = rem >> 4, j = rem & 15;
  int lane = half * 16 + lp;
  int npt = HW >> 4;
  int nkc = Cpad >> 5;
  size_t off = ((((size_t)b * npt + pt) * nkc + kc) * 32 + lane) * 16 + j;
  out[off] = (_Float16)acc;
}

// ---------------------------------------------------------------------------
// Repack pointwise weights (Cout x Cin, f32) into WMMA-A-fragment packed f16,
// zero-padded to [n_ot x 16] x Cpad.
// Layout (halves): [o_tile][k_chunk][lane 0..31][j 0..15]
//   M = lane & 15;  K_in_chunk = j + (lane>=16)*16
// ---------------------------------------------------------------------------
__global__ void k_pack_w(_Float16* __restrict__ ap, const float* __restrict__ w,
                         int Cin, int Cout, int Cpad, int n_ot) {
  int nkc = Cpad >> 5;
  int idx = blockIdx.x * blockDim.x + threadIdx.x;
  int total = n_ot * nkc * 512;
  if (idx >= total) return;
  int j = idx & 15;
  int lane = (idx >> 4) & 31;
  int kc = (idx >> 9) % nkc;
  int ot = idx / (nkc * 512);
  int half = lane >> 4, lr = lane & 15;
  int o = ot * 16 + lr;
  int k = kc * 32 + j + half * 16;
  float v = (o < Cout && k < Cin) ? w[(size_t)o * Cin + k] : 0.0f;
  ap[idx] = (_Float16)v;
}

// ---------------------------------------------------------------------------
// Pointwise conv as WMMA GEMM on pre-packed fragments:
//   out[b,o,p] = sum_c w[o,c] * in[b,c,p] + bias[o]
// One wave per 16(out) x 16(pixel) tile. Inner loop: two v16h loads + 1 WMMA.
// grid: x = ceil(HW/16/4) (4 waves/block), y = n_ot, z = BATCH.
// ---------------------------------------------------------------------------
__global__ void k_pw_wmma(float* __restrict__ out, const _Float16* __restrict__ bp,
                          const _Float16* __restrict__ ap, const float* __restrict__ bias,
                          int Cpad, int Cout, int HW) {
  int wave = threadIdx.x >> 5;
  int lane = threadIdx.x & 31;
  int pt = blockIdx.x * 4 + wave;
  if (pt * 16 >= HW) return;               // wave-uniform
  int ot = blockIdx.y;
  int b  = blockIdx.z;
  int nkc = Cpad >> 5;
  int npt = HW >> 4;
  const v16h* abase = reinterpret_cast<const v16h*>(ap) + ((size_t)ot * nkc) * 32 + lane;
  const v16h* bbase = reinterpret_cast<const v16h*>(bp) + (((size_t)b * npt + pt) * nkc) * 32 + lane;
  v8f acc = {};
#pragma unroll 2
  for (int kc = 0; kc < nkc; ++kc) {
    __builtin_prefetch((const void*)(bbase + (size_t)(kc + 1) * 32), 0, 0);
    v16h afrag = abase[(size_t)kc * 32];
    v16h bfrag = bbase[(size_t)kc * 32];
    acc = __builtin_amdgcn_wmma_f32_16x16x32_f16(
        false, afrag, false, bfrag, (short)0, acc, false, false);
  }
  int half = lane >> 4, lr = lane & 15;
#pragma unroll
  for (int v = 0; v < 8; ++v) {
    int o = ot * 16 + v + half * 8;
    if (o < Cout) out[((size_t)b * Cout + o) * HW + pt * 16 + lr] = acc[v] + bias[o];
  }
}

// ---------------------------------------------------------------------------
// SE: per (b,c) spatial mean. One block per (b,c).
// ---------------------------------------------------------------------------
__global__ void k_se_reduce(float* __restrict__ s, const float* __restrict__ y, int HW) {
  __shared__ float sm[256];
  int bc = blockIdx.x;
  const float* yp = y + (size_t)bc * HW;
  float acc = 0.0f;
  for (int i = threadIdx.x; i < HW; i += 256) acc += yp[i];
  sm[threadIdx.x] = acc;
  __syncthreads();
  for (int st = 128; st > 0; st >>= 1) {
    if (threadIdx.x < st) sm[threadIdx.x] += sm[threadIdx.x + st];
    __syncthreads();
  }
  if (threadIdx.x == 0) s[bc] = sm[0] / (float)HW;
}

// ---------------------------------------------------------------------------
// SE MLP: s1 = relu(s @ w1^T + b1); s2 = sigmoid(s1 @ w2^T + b2). Tiny.
// ---------------------------------------------------------------------------
__global__ void k_se_mlp(float* __restrict__ s2, const float* __restrict__ s,
                         const float* __restrict__ w1, const float* __restrict__ b1,
                         const float* __restrict__ w2, const float* __restrict__ b2,
                         int C, int r) {
  __shared__ float s1[BATCH * 16];
  int t = threadIdx.x;
  if (t < BATCH * r) {
    int b = t / r, u = t % r;
    float acc = b1[u];
    for (int c = 0; c < C; ++c) acc += s[b * C + c] * w1[u * C + c];
    s1[b * 16 + u] = acc > 0.0f ? acc : 0.0f;
  }
  __syncthreads();
  if (t < BATCH * C) {
    int b = t / C, o = t % C;
    float acc = b2[o];
    for (int u = 0; u < r; ++u) acc += s1[b * 16 + u] * w2[o * r + u];
    s2[b * C + o] = 1.0f / (1.0f + expf(-acc));
  }
}

__global__ void k_scale_act(float* __restrict__ y, const float* __restrict__ s2,
                            int C, int HW, int leaky) {
  int idx = blockIdx.x * blockDim.x + threadIdx.x;
  int total = BATCH * C * HW;
  if (idx >= total) return;
  int c = (idx / HW) % C;
  int b = idx / (HW * C);
  float v = y[idx] * s2[b * C + c];
  if (leaky) v = v > 0.0f ? v : 0.1f * v;
  y[idx] = v;
}

// flow offsets (B,2,H,W) -> normalized grid (B,2,H,W)
__global__ void k_apply_offset(float* __restrict__ g, const float* __restrict__ off,
                               int H, int W) {
  int idx = blockIdx.x * blockDim.x + threadIdx.x;
  int HW = H * W;
  if (idx >= BATCH * HW) return;
  int x = idx % W;
  int y = (idx / W) % H;
  int b = idx / HW;
  float gx = (float)x + off[(b * 2 + 0) * HW + y * W + x];
  float gy = (float)y + off[(b * 2 + 1) * HW + y * W + x];
  g[(b * 2 + 0) * HW + y * W + x] = gx / ((float)(W - 1) * 0.5f) - 1.0f;
  g[(b * 2 + 1) * HW + y * W + x] = gy / ((float)(H - 1) * 0.5f) - 1.0f;
}

// bilinear 2x upsample (half-pixel centers, edge clamp), C=2
__global__ void k_upsample2x(float* __restrict__ out, const float* __restrict__ in,
                             int H, int W) {
  int OH = 2 * H, OW = 2 * W;
  int idx = blockIdx.x * blockDim.x + threadIdx.x;
  int total = BATCH * 2 * OH * OW;
  if (idx >= total) return;
  int x = idx % OW;
  int y = (idx / OW) % OH;
  int c = (idx / (OW * OH)) % 2;
  int b = idx / (OW * OH * 2);
  float sx = ((float)x + 0.5f) * 0.5f - 0.5f;
  float sy = ((float)y + 0.5f) * 0.5f - 0.5f;
  sx = fminf(fmaxf(sx, 0.0f), (float)(W - 1));
  sy = fminf(fmaxf(sy, 0.0f), (float)(H - 1));
  int x0 = (int)floorf(sx), y0 = (int)floorf(sy);
  int x1 = x0 + 1 < W - 1 ? x0 + 1 : W - 1;
  int y1 = y0 + 1 < H - 1 ? y0 + 1 : H - 1;
  float wx = sx - (float)x0, wy = sy - (float)y0;
  const float* ip = in + (size_t)(b * 2 + c) * H * W;
  float v00 = ip[y0 * W + x0], v01 = ip[y0 * W + x1];
  float v10 = ip[y1 * W + x0], v11 = ip[y1 * W + x1];
  out[idx] = (v00 * (1 - wx) + v01 * wx) * (1 - wy) + (v10 * (1 - wx) + v11 * wx) * wy;
}

__global__ void k_copy(float* __restrict__ dst, const float* __restrict__ src, int n) {
  int idx = blockIdx.x * blockDim.x + threadIdx.x;
  if (idx < n) dst[idx] = src[idx];
}

// ---------------------------------------------------------------------------
// Host-side orchestration
// ---------------------------------------------------------------------------
// Param leaf order per dsc block (jax tree-flatten, alphabetical):
// 0 dw_b, 1 dw_w, 2 pw_b, 3 pw_w, 4 se1_b, 5 se1_w, 6 se2_b, 7 se2_w
static inline const float* PRM(void* const* d_in, int lev, int net, int layer, int leaf) {
  return (const float*)d_in[11 + (((lev * 2 + net) * 4 + layer) * 8) + leaf];
}

static void run_dsc_net(void* const* d_in, int lev, int net,
                        const float* in0, const float* in1, int csplit,
                        int h, _Float16* bufBp, _Float16* bufAp, float* bufPw,
                        float* sBuf, float* s2Buf, hipStream_t stream) {
  static const int ch_main[5] = {49, 128, 64, 32, 2};
  static const int ch_ref[5]  = {512, 128, 64, 32, 2};
  const int* ch = (net == 0) ? ch_main : ch_ref;
  int HW = h * h;
  const float* src0 = in0;
  const float* src1 = in1 ? in1 : in0;
  int split = csplit;
  for (int layer = 0; layer < 4; ++layer) {
    int Cin = ch[layer], Cout = ch[layer + 1];
    int Cpad = (Cin + 31) & ~31;
    int n_ot = (Cout + 15) / 16;
    int nkc = Cpad >> 5;
    // depthwise conv + pack B fragments (f16, zero K-pad)
    {
      int tot = BATCH * Cpad * HW;
      k_dwconv_pack<<<(tot + 255) / 256, 256, 0, stream>>>(
          bufBp, src0, src1, split, Cin, Cpad,
          PRM(d_in, lev, net, layer, 1), PRM(d_in, lev, net, layer, 0), h, h);
    }
    // pack pointwise weights as A fragments
    {
      int tot = n_ot * nkc * 512;
      k_pack_w<<<(tot + 255) / 256, 256, 0, stream>>>(
          bufAp, PRM(d_in, lev, net, layer, 3), Cin, Cout, Cpad, n_ot);
    }
    // WMMA GEMM + bias
    {
      dim3 grid((HW / 16 + 3) / 4, n_ot, BATCH);
      k_pw_wmma<<<grid, 128, 0, stream>>>(
          bufPw, bufBp, bufAp, PRM(d_in, lev, net, layer, 2), Cpad, Cout, HW);
    }
    // SE block + scale (+ leaky for non-final layers)
    k_se_reduce<<<BATCH * Cout, 256, 0, stream>>>(sBuf, bufPw, HW);
    int r = Cout / 16; if (r < 1) r = 1;
    k_se_mlp<<<1, 256, 0, stream>>>(
        s2Buf, sBuf, PRM(d_in, lev, net, layer, 5), PRM(d_in, lev, net, layer, 4),
        PRM(d_in, lev, net, layer, 7), PRM(d_in, lev, net, layer, 6), Cout, r);
    {
      int tot = BATCH * Cout * HW;
      k_scale_act<<<(tot + 255) / 256, 256, 0, stream>>>(
          bufPw, s2Buf, Cout, HW, layer < 3 ? 1 : 0);
    }
    src0 = bufPw; src1 = bufPw; split = Cout;
  }
}

extern "C" void kernel_launch(void* const* d_in, const int* in_sizes, int n_in,
                              void* d_out, int out_size, void* d_ws, size_t ws_size,
                              hipStream_t stream) {
  (void)in_sizes; (void)n_in; (void)out_size; (void)ws_size;
  float* ws = (float*)d_ws;
  // workspace layout (float offsets), max sizes at h=128
  size_t off_xwa  = 0;                                          // B*256*128*128 f32
  size_t off_bp   = off_xwa  + (size_t)BATCH * 256 * 128 * 128;  // B*512*128*128 f16 (uses half)
  size_t off_pw   = off_bp   + (size_t)BATCH * 256 * 128 * 128;  // B*128*128*128 f32
  size_t off_corr = off_pw   + (size_t)BATCH * 128 * 128 * 128;  // B*49*128*128 f32
  size_t off_fA   = off_corr + (size_t)BATCH * 49 * 128 * 128;   // B*2*256*256 f32
  size_t off_fB   = off_fA   + (size_t)BATCH * 2 * 256 * 256;
  size_t off_fC   = off_fB   + (size_t)BATCH * 2 * 256 * 256;
  size_t off_ap   = off_fC   + (size_t)BATCH * 2 * 256 * 256;    // 64K f16 weight frags
  size_t off_s    = off_ap   + 65536;                            // B*128
  size_t off_s2   = off_s    + (size_t)BATCH * 128;              // B*128
  float*    bufXwa  = ws + off_xwa;
  _Float16* bufBp   = (_Float16*)(ws + off_bp);
  float*    bufPw   = ws + off_pw;
  float*    bufCorr = ws + off_corr;
  _Float16* bufAp   = (_Float16*)(ws + off_ap);
  float*    sBuf    = ws + off_s;
  float*    s2Buf   = ws + off_s2;

  const float* x = (const float*)d_in[0];
  float* L = ws + off_fA;   // last_flow
  float* G = ws + off_fB;   // grid scratch
  float* T = ws + off_fC;   // flow scratch
  bool has = false;

  for (int i = 0; i < 5; ++i) {
    int h = 8 << i;
    int HW = h * h;
    int lev = i;
    const float* xw = (const float*)d_in[1 + 2 * (4 - i)];
    const float* xc = (const float*)d_in[2 + 2 * (4 - i)];

    // 1) warp xw by last_flow (if any)
    const float* xwa = xw;
    if (has) {
      int tot = BATCH * FPN * HW;
      k_grid_sample<<<(tot + 255) / 256, 256, 0, stream>>>(bufXwa, xw, L, FPN, h, h);
      xwa = bufXwa;
    }
    // 2) correlation + leaky
    {
      int tot = BATCH * HW;
      k_corr_leaky<<<(tot + 63) / 64, 64, 0, stream>>>(bufCorr, xwa, xc, h, h);
    }
    // 3) main conv net -> flow_m (B,2,h,h) in bufPw
    run_dsc_net(d_in, lev, 0, bufCorr, nullptr, 49, h, bufBp, bufAp, bufPw, sBuf, s2Buf, stream);
    // 4) fg = apply_offset(flow_m) -> G
    {
      int tot = BATCH * HW;
      k_apply_offset<<<(tot + 255) / 256, 256, 0, stream>>>(G, bufPw, h, h);
    }
    // 5) compose with last_flow
    if (has) {
      int tot = BATCH * 2 * HW;
      k_grid_sample<<<(tot + 255) / 256, 256, 0, stream>>>(T, L, G, 2, h, h);
      float* tmp = L; L = T; T = tmp;      // L := flow; old L becomes scratch
    } else {
      float* tmp = L; L = G; G = tmp;      // L := fg
      has = true;
    }
    // 7) xw2 = warp(xw, L) -> bufXwa
    {
      int tot = BATCH * FPN * HW;
      k_grid_sample<<<(tot + 255) / 256, 256, 0, stream>>>(bufXwa, xw, L, FPN, h, h);
    }
    // 8-9) refine net on concat(xw2, xc) -> flow_r in bufPw
    run_dsc_net(d_in, lev, 1, bufXwa, xc, 256, h, bufBp, bufAp, bufPw, sBuf, s2Buf, stream);
    // 10) fg2 -> G
    {
      int tot = BATCH * HW;
      k_apply_offset<<<(tot + 255) / 256, 256, 0, stream>>>(G, bufPw, h, h);
    }
    // 11) flow2 = grid_sample(L, G) -> T
    {
      int tot = BATCH * 2 * HW;
      k_grid_sample<<<(tot + 255) / 256, 256, 0, stream>>>(T, L, G, 2, h, h);
    }
    // 12) last_flow = upsample2x(T) -> L
    {
      int tot = BATCH * 2 * (2 * h) * (2 * h);
      k_upsample2x<<<(tot + 255) / 256, 256, 0, stream>>>(L, T, h, h);
    }
  }

  // Final: x_warp = grid_sample(x, last_flow); outputs concat [x_warp, last_flow]
  float* outp = (float*)d_out;
  {
    int tot = BATCH * 3 * IMG * IMG;
    k_grid_sample<<<(tot + 255) / 256, 256, 0, stream>>>(outp, x, L, 3, IMG, IMG);
  }
  {
    int n = BATCH * 2 * IMG * IMG;
    k_copy<<<(n + 255) / 256, 256, 0, stream>>>(outp + (size_t)BATCH * 3 * IMG * IMG, L, n);
  }
}
